// CausalSelfAttention_69337952027177
// MI455X (gfx1250) — matrix operand order repaired and checked by reading, not compiled, not run on hardware
//
#include <hip/hip_runtime.h>

// ---------------------------------------------------------------------------
// CDNA5 / gfx1250 causal self-attention forward.
//   phase 1: qkv = x @ Wqkv           (bf16 WMMA, f32 accum)
//   phase 2: flash attention           (bf16 WMMA QK^T and PV, f32 softmax)
//   phase 3: y   = att_out @ Wproj     (bf16 WMMA, f32 accum)
// Outputs (tuple): y (B,T,C) fp32 | k (B,nh,T,hd) fp32 | v (B,nh,T,hd) fp32
//
// All fragments are two contiguous 16B LDS loads (ds_load_b128); B-operands
// are staged N-major in LDS so no strided gathers are needed.  Row-major
// tile staging uses GLOBAL_LOAD_ASYNC_TO_LDS_B128 (ASYNCcnt-tracked).
// ---------------------------------------------------------------------------

typedef __attribute__((ext_vector_type(16))) __bf16 v16bf;
typedef __attribute__((ext_vector_type(8)))  float  v8f;
typedef __attribute__((ext_vector_type(4)))  int    v4i;

#define NEGBIG (-1.0e30f)

constexpr int BATCH = 4;
constexpr int SEQ   = 2048;
constexpr int CH    = 1024;
constexpr int NH    = 16;
constexpr int HD    = 64;
constexpr int MROWS = BATCH * SEQ;   // 8192
constexpr int N_QKV = 3 * CH;        // 3072

#if defined(__gfx1250__)
#if __has_builtin(__builtin_amdgcn_global_load_async_to_lds_b128)
#define HAVE_ASYNC_COPY 1
#endif
#endif

#ifdef HAVE_ASYNC_COPY
typedef __attribute__((address_space(1))) v4i gas_v4i;  // global int4
typedef __attribute__((address_space(3))) v4i las_v4i;  // LDS int4

__device__ __forceinline__ void async_copy16(const void* g, void* l) {
  __builtin_amdgcn_global_load_async_to_lds_b128((gas_v4i*)g, (las_v4i*)l, 0,
                                                 0);
}
__device__ __forceinline__ void async_wait0() {
#if __has_builtin(__builtin_amdgcn_s_wait_asynccnt)
  __builtin_amdgcn_s_wait_asynccnt(0);
#else
  asm volatile("s_wait_asynccnt 0" ::: "memory");
#endif
}
#endif

__device__ __forceinline__ v8f wmma_bf16(v16bf a, v16bf b, v8f c) {
  // (neg_a, A, neg_b, B, c_mod, C, reuse_a, reuse_b)
  return __builtin_amdgcn_wmma_f32_16x16x32_bf16(false, a, false, b, (short)0,
                                                 c, false, false);
}

union U16bf { v16bf v; uint4 q[2]; };

// A fragment: 16x32 (MxK) bf16 from row-major source (ld bf16 elements).
// ISA layout -> per lane: two contiguous 8-element runs at K = klo, klo+16.
__device__ __forceinline__ v16bf frag_a(const __bf16* p, int ld, int lane) {
  const int m   = lane & 15;
  const int klo = (lane & 16) ? 8 : 0;
  U16bf u;
  u.q[0] = *(const uint4*)(p + m * ld + klo);
  u.q[1] = *(const uint4*)(p + m * ld + klo + 16);
  return u.v;
}

// B fragment: 32x16 (KxN) bf16 from an N-major source: B[k][n] = p[n*ld + k].
// Per lane: two contiguous 8-element runs at K = kb, kb+16.
__device__ __forceinline__ v16bf frag_bn(const __bf16* p, int ld, int lane) {
  const int n  = lane & 15;
  const int kb = (lane & 16) ? 8 : 0;
  U16bf u;
  u.q[0] = *(const uint4*)(p + n * ld + kb);
  u.q[1] = *(const uint4*)(p + n * ld + kb + 16);
  return u.v;
}

// ---------------------------------------------------------------------------
// Phase 1: qkv GEMM.  M=8192, N=3072, K=1024.  Block tile 128x64, 8 waves,
// each wave a 32x32 tile (2x2 WMMA frags), K stepped by 32 through LDS.
// ---------------------------------------------------------------------------
__global__ __launch_bounds__(256) void qkv_gemm_kernel(
    const float* __restrict__ x, const float* __restrict__ Wqkv,
    float* __restrict__ k_out, float* __restrict__ v_out,
    __bf16* __restrict__ q_bf, __bf16* __restrict__ k_bf,
    __bf16* __restrict__ v_bf) {
  __shared__ __align__(16) __bf16 As[128][40];   // 128x32 row-major (+8 pad)
  __shared__ __align__(16) __bf16 BsT[64][40];   // 64(N) x 32(K) N-major
  const int tid  = threadIdx.x;
  const int lane = tid & 31;
  const int wave = tid >> 5;
  const int m0 = blockIdx.x * 128;
  const int n0 = blockIdx.y * 64;
  const int wm = (wave & 3) * 32;
  const int wn = (wave >> 2) * 32;

  v8f acc[2][2] = {};

  for (int k0 = 0; k0 < CH; k0 += 32) {
#pragma unroll
    for (int i = 0; i < 4; ++i) {               // A: 128x32 f32 -> bf16
      const int idx = tid + i * 256;            // 1024 float4
      const int r = idx >> 3, c4 = (idx & 7) * 4;
      const float4 f = *(const float4*)(x + (size_t)(m0 + r) * CH + k0 + c4);
      As[r][c4 + 0] = (__bf16)f.x; As[r][c4 + 1] = (__bf16)f.y;
      As[r][c4 + 2] = (__bf16)f.z; As[r][c4 + 3] = (__bf16)f.w;
    }
#pragma unroll
    for (int i = 0; i < 2; ++i) {               // B: 32x64 f32 -> bf16, transposed
      const int idx = tid + i * 256;            // 512 float4
      const int r = idx >> 4, c4 = (idx & 15) * 4;
      const float4 f =
          *(const float4*)(Wqkv + (size_t)(k0 + r) * N_QKV + n0 + c4);
      BsT[c4 + 0][r] = (__bf16)f.x; BsT[c4 + 1][r] = (__bf16)f.y;
      BsT[c4 + 2][r] = (__bf16)f.z; BsT[c4 + 3][r] = (__bf16)f.w;
    }
    __syncthreads();
    const v16bf a0 = frag_a(&As[wm][0],       40, lane);
    const v16bf a1 = frag_a(&As[wm + 16][0],  40, lane);
    const v16bf b0 = frag_bn(&BsT[wn][0],      40, lane);
    const v16bf b1 = frag_bn(&BsT[wn + 16][0], 40, lane);
    acc[0][0] = wmma_bf16(a0, b0, acc[0][0]);
    acc[0][1] = wmma_bf16(a0, b1, acc[0][1]);
    acc[1][0] = wmma_bf16(a1, b0, acc[1][0]);
    acc[1][1] = wmma_bf16(a1, b1, acc[1][1]);
    __syncthreads();
  }

  const int nlo = lane & 15;
  const int mh  = (lane & 16) ? 8 : 0;
#pragma unroll
  for (int fm = 0; fm < 2; ++fm)
#pragma unroll
    for (int fn = 0; fn < 2; ++fn)
#pragma unroll
      for (int r = 0; r < 8; ++r) {
        const int row = m0 + wm + fm * 16 + mh + r;
        const int col = n0 + wn + fn * 16 + nlo;
        const float val = acc[fm][fn][r];
        const int b = row >> 11;          // row / SEQ
        const int t = row & (SEQ - 1);
        if (col < CH) {                   // q -> bf16 workspace (B,nh,T,hd)
          const int h = col >> 6, d = col & 63;
          q_bf[(((size_t)b * NH + h) * SEQ + t) * HD + d] = (__bf16)val;
        } else if (col < 2 * CH) {        // k -> output + bf16 workspace
          const int c = col - CH, h = c >> 6, d = c & 63;
          const size_t idx = (((size_t)b * NH + h) * SEQ + t) * HD + d;
          k_out[idx] = val;
          k_bf[idx]  = (__bf16)val;
        } else {                          // v -> output + bf16 workspace
          const int c = col - 2 * CH, h = c >> 6, d = c & 63;
          const size_t idx = (((size_t)b * NH + h) * SEQ + t) * HD + d;
          v_out[idx] = val;
          v_bf[idx]  = (__bf16)val;
        }
      }
}

// ---------------------------------------------------------------------------
// Phase 2: flash attention.  Grid (B*nh, T/64); 4 waves/WG; wave owns 16 query
// rows and the full 16x64 O accumulator.  Keys stream in 32-wide blocks
// through LDS shared by the workgroup; online softmax in f32 registers.
// ---------------------------------------------------------------------------
__global__ __launch_bounds__(128) void flash_attn_kernel(
    const __bf16* __restrict__ q_bf, const __bf16* __restrict__ k_bf,
    const __bf16* __restrict__ v_bf, __bf16* __restrict__ y_att) {
  __shared__ __align__(16) __bf16 Qs[64][72];     // 64x64 query tile (+8 pad)
  __shared__ __align__(16) __bf16 Kt[32][64];     // 32 keys x hd, row-major
  __shared__ __align__(16) __bf16 VtT[64][40];    // hd x 32 keys, d-major
  __shared__ __align__(16) __bf16 Pw[4][16][32];  // per-wave P re-layout

  const int tid  = threadIdx.x;
  const int lane = tid & 31;
  const int wave = tid >> 5;
  const int bh = blockIdx.x;                  // 0..63
  const int b  = bh >> 4, h = bh & 15;
  const int q0wg = blockIdx.y * 64;
  const int q0   = q0wg + wave * 16;
  const size_t base = (size_t)bh * SEQ * HD;  // into (B,nh,T,hd) bf16 arrays

  // Stage the 64x64 Q tile (row-major, 16B chunks -> async path if available).
#pragma unroll
  for (int i = 0; i < 4; ++i) {
    const int idx = tid + i * 128;            // 512 x 16B
    const int r = idx >> 3, c8 = (idx & 7) * 8;
    const __bf16* src = q_bf + base + (size_t)(q0wg + r) * HD + c8;
#ifdef HAVE_ASYNC_COPY
    async_copy16(src, &Qs[r][c8]);
#else
    *(uint4*)&Qs[r][c8] = *(const uint4*)src;
#endif
  }
#ifdef HAVE_ASYNC_COPY
  async_wait0();
#endif
  __syncthreads();
  const v16bf qa0 = frag_a(&Qs[wave * 16][0],  72, lane);
  const v16bf qa1 = frag_a(&Qs[wave * 16][32], 72, lane);

  v8f o[4] = {};
  float mrow[8], lrow[8];
#pragma unroll
  for (int r = 0; r < 8; ++r) { mrow[r] = NEGBIG; lrow[r] = 0.0f; }

  const int nlo = lane & 15;
  const int mh  = (lane & 16) ? 8 : 0;
  const int nkb = q0wg / 32 + 2;              // causal block count for the WG

  for (int kb = 0; kb < nkb; ++kb) {
    const int k0 = kb * 32;
    __syncthreads();                          // previous-block readers done
#pragma unroll
    for (int i = 0; i < 2; ++i) {             // K: row-major 32x64 stage
      const int idx = tid + i * 128;          // 256 x 16B
      const int r = idx >> 3, c8 = (idx & 7) * 8;
      const __bf16* src = k_bf + base + (size_t)(k0 + r) * HD + c8;
#ifdef HAVE_ASYNC_COPY
      async_copy16(src, &Kt[r][c8]);
#else
      *(uint4*)&Kt[r][c8] = *(const uint4*)src;
#endif
    }
#pragma unroll
    for (int i = 0; i < 2; ++i) {             // V: stage transposed (d-major)
      const int idx = tid + i * 128;
      const int r = idx >> 3, c8 = (idx & 7) * 8;
      const uint4 fv =
          *(const uint4*)(v_bf + base + (size_t)(k0 + r) * HD + c8);
      const __bf16* pv = (const __bf16*)&fv;
#pragma unroll
      for (int j = 0; j < 8; ++j) VtT[c8 + j][r] = pv[j];
    }
    if (k0 + 32 < SEQ) {                      // global_prefetch next block
      __builtin_prefetch(k_bf + base + (size_t)(k0 + 32) * HD, 0, 1);
      __builtin_prefetch(v_bf + base + (size_t)(k0 + 32) * HD, 0, 1);
    }
#ifdef HAVE_ASYNC_COPY
    async_wait0();
#endif
    __syncthreads();
    if (k0 > q0 + 15) continue;               // causally dead for this wave

    // S(16x32) = Q(16x64) . K^T : B[k=hd][n=key] = Kt[key][hd] (N-major view)
    v8f s0 = {}, s1 = {};
    s0 = wmma_bf16(qa0, frag_bn(&Kt[0][0],   HD, lane), s0);
    s0 = wmma_bf16(qa1, frag_bn(&Kt[0][32],  HD, lane), s0);
    s1 = wmma_bf16(qa0, frag_bn(&Kt[16][0],  HD, lane), s1);
    s1 = wmma_bf16(qa1, frag_bn(&Kt[16][32], HD, lane), s1);

    // Online softmax; xor masks 1..8 stay inside each 16-lane half, matching
    // the C-frag row ownership (row M lives in one VGPR across 16 lanes).
#pragma unroll
    for (int r = 0; r < 8; ++r) {
      float a = s0[r] * 0.125f;               // 1/sqrt(64)
      float c = s1[r] * 0.125f;
      const int qrow = q0 + mh + r;
      if (k0 + nlo > qrow)      a = NEGBIG;   // causal mask
      if (k0 + 16 + nlo > qrow) c = NEGBIG;
      float mx = fmaxf(a, c);
#pragma unroll
      for (int off = 8; off >= 1; off >>= 1)
        mx = fmaxf(mx, __shfl_xor(mx, off, 32));
      const float mnew = fmaxf(mrow[r], mx);
      const float corr = __expf(mrow[r] - mnew);
      const float p0 = __expf(a - mnew);
      const float p1 = __expf(c - mnew);
      mrow[r] = mnew;
      Pw[wave][mh + r][nlo]      = (__bf16)p0;
      Pw[wave][mh + r][16 + nlo] = (__bf16)p1;
      float ps = p0 + p1;
#pragma unroll
      for (int off = 8; off >= 1; off >>= 1)
        ps += __shfl_xor(ps, off, 32);
      lrow[r] = lrow[r] * corr + ps;
#pragma unroll
      for (int cf = 0; cf < 4; ++cf) o[cf][r] *= corr;
    }

    // O(16x64) += P(16x32) . V(32x64).  B[k=key][n=d] = VtT[d][key] (N-major).
    // Pw store->load is same-wave LDS (in-order; compiler inserts dscnt waits).
    const v16bf pa = frag_a(&Pw[wave][0][0], 32, lane);
#pragma unroll
    for (int cf = 0; cf < 4; ++cf)
      o[cf] = wmma_bf16(pa, frag_bn(&VtT[cf * 16][0], 40, lane), o[cf]);
  }

  // Normalize and write attention output as bf16 (B,T,C) for the proj GEMM.
  float inv[8];
#pragma unroll
  for (int r = 0; r < 8; ++r) inv[r] = 1.0f / lrow[r];
#pragma unroll
  for (int cf = 0; cf < 4; ++cf) {
    const int d = h * HD + cf * 16 + nlo;
#pragma unroll
    for (int r = 0; r < 8; ++r) {
      const int t = q0 + mh + r;
      y_att[((size_t)b * SEQ + t) * CH + d] = (__bf16)(o[cf][r] * inv[r]);
    }
  }
}

// ---------------------------------------------------------------------------
// Phase 3: output projection.  M=8192, N=1024, K=1024.  Same tiling as qkv;
// the bf16 A tile is a pure row-major copy -> async LDS path if available.
// ---------------------------------------------------------------------------
__global__ __launch_bounds__(256) void proj_gemm_kernel(
    const __bf16* __restrict__ y_att, const float* __restrict__ Wproj,
    float* __restrict__ y_out) {
  __shared__ __align__(16) __bf16 As[128][40];
  __shared__ __align__(16) __bf16 BsT[64][40];
  const int tid  = threadIdx.x;
  const int lane = tid & 31;
  const int wave = tid >> 5;
  const int m0 = blockIdx.x * 128;
  const int n0 = blockIdx.y * 64;
  const int wm = (wave & 3) * 32;
  const int wn = (wave >> 2) * 32;

  v8f acc[2][2] = {};

  for (int k0 = 0; k0 < CH; k0 += 32) {
#pragma unroll
    for (int i = 0; i < 2; ++i) {             // A: 128x32 bf16 row-major copy
      const int idx = tid + i * 256;          // 512 x 16B
      const int r = idx >> 2, c8 = (idx & 3) * 8;
      const __bf16* src = y_att + (size_t)(m0 + r) * CH + k0 + c8;
#ifdef HAVE_ASYNC_COPY
      async_copy16(src, &As[r][c8]);
#else
      *(uint4*)&As[r][c8] = *(const uint4*)src;
#endif
    }
#pragma unroll
    for (int i = 0; i < 2; ++i) {             // B: 32x64 f32 -> bf16, transposed
      const int idx = tid + i * 256;
      const int r = idx >> 4, c4 = (idx & 15) * 4;
      const float4 f =
          *(const float4*)(Wproj + (size_t)(k0 + r) * CH + n0 + c4);
      BsT[c4 + 0][r] = (__bf16)f.x; BsT[c4 + 1][r] = (__bf16)f.y;
      BsT[c4 + 2][r] = (__bf16)f.z; BsT[c4 + 3][r] = (__bf16)f.w;
    }
#ifdef HAVE_ASYNC_COPY
    async_wait0();
#endif
    __syncthreads();
    const v16bf a0 = frag_a(&As[wm][0],       40, lane);
    const v16bf a1 = frag_a(&As[wm + 16][0],  40, lane);
    const v16bf b0 = frag_bn(&BsT[wn][0],      40, lane);
    const v16bf b1 = frag_bn(&BsT[wn + 16][0], 40, lane);
    acc[0][0] = wmma_bf16(a0, b0, acc[0][0]);
    acc[0][1] = wmma_bf16(a0, b1, acc[0][1]);
    acc[1][0] = wmma_bf16(a1, b0, acc[1][0]);
    acc[1][1] = wmma_bf16(a1, b1, acc[1][1]);
    __syncthreads();
  }

  const int nlo = lane & 15;
  const int mh  = (lane & 16) ? 8 : 0;
#pragma unroll
  for (int fm = 0; fm < 2; ++fm)
#pragma unroll
    for (int fn = 0; fn < 2; ++fn)
#pragma unroll
      for (int r = 0; r < 8; ++r) {
        const int row = m0 + wm + fm * 16 + mh + r;
        const int col = n0 + wn + fn * 16 + nlo;
        y_out[(size_t)row * CH + col] = acc[fm][fn][r];
      }
}

// ---------------------------------------------------------------------------
extern "C" void kernel_launch(void* const* d_in, const int* in_sizes, int n_in,
                              void* d_out, int out_size, void* d_ws,
                              size_t ws_size, hipStream_t stream) {
  (void)in_sizes; (void)n_in; (void)out_size; (void)ws_size;
  const float* x     = (const float*)d_in[0];
  const float* Wqkv  = (const float*)d_in[1];
  const float* Wproj = (const float*)d_in[2];

  float* y     = (float*)d_out;                       // (B,T,C)
  float* k_out = y + (size_t)BATCH * SEQ * CH;        // (B,nh,T,hd)
  float* v_out = k_out + (size_t)BATCH * SEQ * CH;    // (B,nh,T,hd)

  const size_t elems = (size_t)BATCH * SEQ * CH;      // 8M elements
  char* ws = (char*)d_ws;
  __bf16* q_bf  = (__bf16*)(ws);                      // 16 MB
  __bf16* k_bf  = (__bf16*)(ws + 2 * elems);          // 16 MB
  __bf16* v_bf  = (__bf16*)(ws + 4 * elems);          // 16 MB
  __bf16* y_att = (__bf16*)(ws + 6 * elems);          // 16 MB

  qkv_gemm_kernel<<<dim3(MROWS / 128, N_QKV / 64), 256, 0, stream>>>(
      x, Wqkv, k_out, v_out, q_bf, k_bf, v_bf);
  flash_attn_kernel<<<dim3(BATCH * NH, SEQ / 64), 128, 0, stream>>>(
      q_bf, k_bf, v_bf, y_att);
  proj_gemm_kernel<<<dim3(MROWS / 128, CH / 64), 256, 0, stream>>>(
      y_att, Wproj, y);
}
